// Attention_46402826666684
// MI455X (gfx1250) — matrix operand order repaired and checked by reading, not compiled, not run on hardware
//
#include <hip/hip_runtime.h>
#include <hip/hip_bf16.h>

#define BB   2
#define SS   2048
#define DD   4096
#define HH   32
#define KVHH 8
#define HDD  128
// N_REP = 4

typedef __bf16 v16bf __attribute__((ext_vector_type(16)));
typedef __bf16 v8bf  __attribute__((ext_vector_type(8)));
typedef __bf16 v4bf  __attribute__((ext_vector_type(4)));
typedef float  v8f   __attribute__((ext_vector_type(8)));
typedef unsigned int u32x4 __attribute__((ext_vector_type(4)));
typedef int          i32x4 __attribute__((ext_vector_type(4)));
typedef int          i32x8 __attribute__((ext_vector_type(8)));

// ---- optional CDNA5 TDM path (guarded so compile never regresses) ----------
#if defined(__has_builtin)
#if __has_builtin(__builtin_amdgcn_tensor_load_to_lds) && __has_builtin(__builtin_amdgcn_s_wait_tensorcnt)
#define USE_TDM 1
#endif
#endif
#if defined(__has_include)
#if __has_include(<hip/amd_detail/amd_gfx1250_TDM.h>)
#define TDM_SIX_ARGS 1
#endif
#endif

static __device__ __forceinline__ __bf16 f2bf(float f) {
    unsigned int u = __builtin_bit_cast(unsigned int, f);
    unsigned int r = (u + 0x7FFFu + ((u >> 16) & 1u)) >> 16;
    return __builtin_bit_cast(__bf16, (unsigned short)r);
}

// A-fragment (16x32 bf16, M x K): lane holds row=lane%16, 16 contiguous K starting at
// kb = chunk + 16*(lane>=16).
static __device__ __forceinline__ v16bf load_a_frag(const __bf16* base, int stride, int row, int kbase) {
    const v8bf lo = *reinterpret_cast<const v8bf*>(base + row * stride + kbase);
    const v8bf hi = *reinterpret_cast<const v8bf*>(base + row * stride + kbase + 8);
    return __builtin_shufflevector(lo, hi, 0,1,2,3,4,5,6,7,8,9,10,11,12,13,14,15);
}

// B-fragment (32x16 bf16, K x N): lane holds col=lane%16, 16 contiguous K starting at
// ks = chunk + 16*(lane>=16). Requires N-major storage (K contiguous per column).
static __device__ __forceinline__ v16bf load_b_frag(const __bf16* base, int stride, int col, int ks) {
    const v8bf lo = *reinterpret_cast<const v8bf*>(base + col * stride + ks);
    const v8bf hi = *reinterpret_cast<const v8bf*>(base + col * stride + ks + 8);
    return __builtin_shufflevector(lo, hi, 0,1,2,3,4,5,6,7,8,9,10,11,12,13,14,15);
}

static __device__ __forceinline__ v8f wmma_bf16(v16bf a, v16bf b, v8f c) {
    return __builtin_amdgcn_wmma_f32_16x16x32_bf16(false, a, false, b, (short)0, c, false, false);
}

#if defined(USE_TDM)
// TDM: 2-D tile load, 128 rows x 128 bf16, memory row stride 128 elems,
// LDS destination padded +8 elems (4 dwords) after every 64 dwords (one row).
static __device__ __forceinline__ void tdm_load_k_tile(unsigned lds_off, const void* gaddr) {
    const unsigned long long ga = (unsigned long long)(size_t)gaddr;
    u32x4 g0;
    g0[0] = 1u;                                             // count=1, user D#
    g0[1] = lds_off;                                        // lds_addr (bytes)
    g0[2] = (unsigned)(ga & 0xFFFFFFFFu);                   // global_addr[31:0]
    g0[3] = (unsigned)((ga >> 32) & 0x1FFFFFFu) | (2u << 30); // global_addr[56:32], type=2
    i32x8 g1;
    g1[0] = (int)((1u << 16) | (1u << 20) | (5u << 22) | (3u << 25)); // 2B elems, pad: 4dw per 64dw
    g1[1] = (int)(128u << 16);      // tensor_dim0 = 128
    g1[2] = (int)(2048u << 16);     // tensor_dim1 = 2048 (lo16)
    g1[3] = (int)(128u << 16);      // tile_dim0 = 128
    g1[4] = (int)128;               // tile_dim1 = 128, tile_dim2 = 0
    g1[5] = (int)128;               // tensor_dim0_stride = 128
    g1[6] = 0; g1[7] = 0;
    i32x4 z4 = {0, 0, 0, 0};
#if defined(TDM_SIX_ARGS)
    i32x8 z8 = {0, 0, 0, 0, 0, 0, 0, 0};
    __builtin_amdgcn_tensor_load_to_lds(g0, g1, z4, z4, z8, 0);
#else
    __builtin_amdgcn_tensor_load_to_lds(g0, g1, z4, z4, 0);
#endif
}
#endif

// ---------------------------------------------------------------------------
// GEMM: C[M,N] = A[M,K] * B[K,N], bf16 WMMA, f32 accumulate.
// 128x128 C tile, K step 32, 256 threads = 8 waves (2x4 16x16 tiles each).
// Double-buffered LDS with register staging: ldg(next) -> wmma(cur) -> sts(next)
// -> single barrier per K step.
// ---------------------------------------------------------------------------
template <bool AF32>
__global__ __launch_bounds__(256) void gemm_k(const void* __restrict__ Ap,
                                              const float* __restrict__ Bw,
                                              float* __restrict__ C,
                                              int M, int N, int K) {
    __shared__ __bf16 Al[2][128 * 40];   // row-major [m][k], pad 8
    __shared__ __bf16 Bl[2][128 * 40];   // N-major  [n][k], pad 8

    const int t = threadIdx.x;
    const int lane = t & 31, wave = t >> 5;
    const int wm = wave & 3, wn = wave >> 2;     // 4 waves in M, 2 in N
    const int half = lane >> 4, nl = lane & 15;
    const int m0 = blockIdx.y * 128, n0 = blockIdx.x * 128;
    const int arow = t >> 1, ac0 = (t & 1) * 16;     // A stage: 1 row-half / thread
    const int bk = t >> 3,  bn1 = (t & 7) * 16;      // B stage: 16 n / thread at one k

    float4 fa[4]; v8bf ab[2]; float4 fb[4];

    auto ldg = [&](int kb) {
        if constexpr (AF32) {
            const float* asrc = (const float*)Ap + (size_t)(m0 + arow) * K + kb + ac0;
#pragma unroll
            for (int v = 0; v < 4; ++v) fa[v] = *reinterpret_cast<const float4*>(asrc + 4 * v);
        } else {
            const __bf16* asrc = (const __bf16*)Ap + (size_t)(m0 + arow) * K + kb + ac0;
            ab[0] = *reinterpret_cast<const v8bf*>(asrc);
            ab[1] = *reinterpret_cast<const v8bf*>(asrc + 8);
        }
        const float* bsrc = Bw + (size_t)(kb + bk) * N + n0 + bn1;
#pragma unroll
        for (int v = 0; v < 4; ++v) fb[v] = *reinterpret_cast<const float4*>(bsrc + 4 * v);
    };
    auto sts = [&](int buf) {
        if constexpr (AF32) {
#pragma unroll
            for (int v = 0; v < 4; ++v) {
                v4bf p;
                p[0] = f2bf(fa[v].x); p[1] = f2bf(fa[v].y);
                p[2] = f2bf(fa[v].z); p[3] = f2bf(fa[v].w);
                *reinterpret_cast<v4bf*>(&Al[buf][arow * 40 + ac0 + v * 4]) = p;
            }
        } else {
            *reinterpret_cast<v8bf*>(&Al[buf][arow * 40 + ac0]) = ab[0];
            *reinterpret_cast<v8bf*>(&Al[buf][arow * 40 + ac0 + 8]) = ab[1];
        }
#pragma unroll
        for (int v = 0; v < 4; ++v) {
            const float* fp = &fb[v].x;
#pragma unroll
            for (int j = 0; j < 4; ++j)
                Bl[buf][(bn1 + v * 4 + j) * 40 + bk] = f2bf(fp[j]);
        }
    };

    v8f acc[2][4] = {};

    ldg(0);
    sts(0);
    __syncthreads();

    int buf = 0;
    for (int kb = 0; kb < K; kb += 32) {
        const int nkb = kb + 32;
        if (nkb < K) {
            ldg(nkb);
            if (nkb + 32 < K) {  // prime L2 one more step ahead
                __builtin_prefetch((const void*)(Bw + (size_t)(nkb + 32 + bk) * N + n0 + bn1), 0, 0);
                if constexpr (AF32)
                    __builtin_prefetch((const void*)((const float*)Ap + (size_t)(m0 + arow) * K + nkb + 32 + ac0), 0, 0);
            }
        }
        v16bf af[2];
#pragma unroll
        for (int i = 0; i < 2; ++i)
            af[i] = load_a_frag(Al[buf], 40, wm * 32 + i * 16 + nl, half * 16);
#pragma unroll
        for (int j = 0; j < 4; ++j) {
            v16bf bf = load_b_frag(Bl[buf], 40, wn * 64 + j * 16 + nl, half * 16);
#pragma unroll
            for (int i = 0; i < 2; ++i)
                acc[i][j] = wmma_bf16(af[i], bf, acc[i][j]);
        }
        if (nkb < K) sts(buf ^ 1);
        __syncthreads();
        buf ^= 1;
    }

    // D layout: lane -> col = lane%16; row = vgpr + 8*(lane>=16)
#pragma unroll
    for (int i = 0; i < 2; ++i)
#pragma unroll
        for (int j = 0; j < 4; ++j)
#pragma unroll
            for (int r = 0; r < 8; ++r)
                C[(size_t)(m0 + wm * 32 + i * 16 + r + 8 * half) * N + n0 + wn * 64 + j * 16 + nl] =
                    acc[i][j][r];
}

// ---------------------------------------------------------------------------
// RoPE + repack Q: q f32 [B,S,H*HD] -> bf16 [B,H,S,HD]
// ---------------------------------------------------------------------------
__global__ __launch_bounds__(256) void rope_q_k(const float* __restrict__ q,
                                                const float* __restrict__ cosb,
                                                const float* __restrict__ sinb,
                                                __bf16* __restrict__ qb) {
    const unsigned idx = blockIdx.x * 256u + threadIdx.x;   // B*S*H*64
    const int i = idx & 63;
    const int h = (idx >> 6) & 31;
    const int s = (idx >> 11) & 2047;
    const int b = idx >> 22;
    const float c = cosb[s * 64 + i], sn = sinb[s * 64 + i];
    const size_t src = ((size_t)(b * SS + s)) * (HH * HDD) + h * HDD + 2 * i;
    const float t0 = q[src], t1 = q[src + 1];
    const size_t dst = (((size_t)(b * HH + h)) * SS + s) * HDD + 2 * i;
    qb[dst]     = f2bf(t0 * c - t1 * sn);
    qb[dst + 1] = f2bf(t0 * sn + t1 * c);
}

// RoPE K + convert V: f32 [B,S,KVH*HD] -> bf16 [B,KVH,S,HD]
__global__ __launch_bounds__(256) void rope_kv_k(const float* __restrict__ k,
                                                 const float* __restrict__ v,
                                                 const float* __restrict__ cosb,
                                                 const float* __restrict__ sinb,
                                                 __bf16* __restrict__ kb,
                                                 __bf16* __restrict__ vb) {
    const unsigned idx = blockIdx.x * 256u + threadIdx.x;   // B*S*KVH*64
    const int i = idx & 63;
    const int hk = (idx >> 6) & 7;
    const int s = (idx >> 9) & 2047;
    const int b = idx >> 20;
    const float c = cosb[s * 64 + i], sn = sinb[s * 64 + i];
    const size_t src = ((size_t)(b * SS + s)) * (KVHH * HDD) + hk * HDD + 2 * i;
    const size_t dst = (((size_t)(b * KVHH + hk)) * SS + s) * HDD + 2 * i;
    const float t0 = k[src], t1 = k[src + 1];
    kb[dst]     = f2bf(t0 * c - t1 * sn);
    kb[dst + 1] = f2bf(t0 * sn + t1 * c);
    vb[dst]     = f2bf(v[src]);
    vb[dst + 1] = f2bf(v[src + 1]);
}

// ---------------------------------------------------------------------------
// Causal flash attention. Grid: (S/128, B*H). 256 threads = 8 waves;
// each wave owns 16 query rows. LDS: Q[128][136], K[128][136], V^T[128][136],
// P[8 waves][16][136], all bf16 (139264 B dynamic).
// K tile staged by the Tensor Data Mover when available (padded-LDS D#),
// fenced with s_wait_tensorcnt; V transposed manually (TDM cannot transpose).
// ---------------------------------------------------------------------------
__global__ __launch_bounds__(256) void attn_k(const __bf16* __restrict__ qb,
                                              const __bf16* __restrict__ kb,
                                              const __bf16* __restrict__ vb,
                                              __bf16* __restrict__ attn) {
    extern __shared__ char smem_raw[];
    __bf16* Ql = reinterpret_cast<__bf16*>(smem_raw);
    __bf16* Kl = Ql + 128 * 136;
    __bf16* Vl = Kl + 128 * 136;   // stored [hd][key]
    __bf16* Pl = Vl + 128 * 136;   // 8 waves x [16][136]

    const int t = threadIdx.x, lane = t & 31, wave = t >> 5;
    const int half = lane >> 4, nl = lane & 15;
    const int qt = blockIdx.x;
    const int bh = blockIdx.y;
    const int b = bh >> 5, h = bh & 31;
    const int hk = h >> 2;                       // N_REP = 4
    const int q0 = qt * 128;
    const float scale = 0.08838834764831845f;    // 1/sqrt(128)
    const int srow = t >> 1, sc0 = (t & 1) * 64; // staging: one row-half / thread

    const __bf16* kbase = kb + ((size_t)(b * KVHH + hk)) * SS * HDD;
    const __bf16* vbase = vb + ((size_t)(b * KVHH + hk)) * SS * HDD;

    // stage Q once
    {
        const __bf16* qsrc = qb + (((size_t)(b * HH + h)) * SS + q0) * HDD;
#pragma unroll
        for (int v = 0; v < 8; ++v)
            *reinterpret_cast<v8bf*>(&Ql[srow * 136 + sc0 + v * 8]) =
                *reinterpret_cast<const v8bf*>(qsrc + (size_t)srow * HDD + sc0 + v * 8);
    }

    v8f oacc[8] = {};
    float mrow[8], lrow[8];
#pragma unroll
    for (int r = 0; r < 8; ++r) { mrow[r] = -1e30f; lrow[r] = 0.0f; }

    for (int it = 0; it <= qt; ++it) {
        const int kc0 = it * 128;
        __syncthreads();
        const __bf16* ksrc = kbase + (size_t)kc0 * HDD;
        const __bf16* vsrc = vbase + (size_t)kc0 * HDD;
#if defined(USE_TDM)
        if (wave == 0)
            tdm_load_k_tile((unsigned)(size_t)Kl, (const void*)ksrc);
#else
#pragma unroll
        for (int v = 0; v < 8; ++v)
            *reinterpret_cast<v8bf*>(&Kl[srow * 136 + sc0 + v * 8]) =
                *reinterpret_cast<const v8bf*>(ksrc + (size_t)srow * HDD + sc0 + v * 8);
#endif
        // V transposed: [hd][key]
#pragma unroll
        for (int v = 0; v < 8; ++v) {
            v8bf d = *reinterpret_cast<const v8bf*>(vsrc + (size_t)srow * HDD + sc0 + v * 8);
#pragma unroll
            for (int j = 0; j < 8; ++j)
                Vl[(sc0 + v * 8 + j) * 136 + srow] = d[j];
        }
        if (it < qt) {  // prime L2 for the next key block
            __builtin_prefetch((const void*)(ksrc + (size_t)128 * HDD + (size_t)srow * HDD + sc0), 0, 0);
            __builtin_prefetch((const void*)(vsrc + (size_t)128 * HDD + (size_t)srow * HDD + sc0), 0, 0);
        }
#if defined(USE_TDM)
        if (wave == 0)
            __builtin_amdgcn_s_wait_tensorcnt(0);
#endif
        __syncthreads();

        // S = Q K^T  (wave's 16 rows x 128 keys)
        v8f sacc[8] = {};
#pragma unroll
        for (int kc = 0; kc < 4; ++kc) {
            v16bf af = load_a_frag(Ql, 136, wave * 16 + nl, kc * 32 + half * 16);
#pragma unroll
            for (int nt = 0; nt < 8; ++nt) {
                v16bf bf = load_b_frag(Kl, 136, nt * 16 + nl, kc * 32 + half * 16);
                sacc[nt] = wmma_bf16(af, bf, sacc[nt]);
            }
        }

        // scale + causal mask + per-row max (row = r + 8*half, col per lane)
        float rm[8];
#pragma unroll
        for (int r = 0; r < 8; ++r) rm[r] = -1e30f;
#pragma unroll
        for (int nt = 0; nt < 8; ++nt) {
            const int col = kc0 + nt * 16 + nl;
#pragma unroll
            for (int r = 0; r < 8; ++r) {
                const int qrow = q0 + wave * 16 + r + 8 * half;
                float s = sacc[nt][r] * scale;
                if (col > qrow) s = -1e30f;
                sacc[nt][r] = s;
                rm[r] = fmaxf(rm[r], s);
            }
        }
#pragma unroll
        for (int off = 1; off < 16; off <<= 1)
#pragma unroll
            for (int r = 0; r < 8; ++r)
                rm[r] = fmaxf(rm[r], __shfl_xor(rm[r], off, 32));

        float alpha[8], rs[8];
#pragma unroll
        for (int r = 0; r < 8; ++r) {
            const float mn = fmaxf(mrow[r], rm[r]);
            alpha[r] = __expf(mrow[r] - mn);
            mrow[r] = mn;
            rs[r] = 0.0f;
        }

        // P = exp(S - m), row sums, stash P (bf16) in this wave's LDS region
        __bf16* Pw = Pl + wave * 16 * 136;
#pragma unroll
        for (int nt = 0; nt < 8; ++nt)
#pragma unroll
            for (int r = 0; r < 8; ++r) {
                const float p = __expf(sacc[nt][r] - mrow[r]);
                rs[r] += p;
                Pw[(r + 8 * half) * 136 + nt * 16 + nl] = f2bf(p);
            }
#pragma unroll
        for (int off = 1; off < 16; off <<= 1)
#pragma unroll
            for (int r = 0; r < 8; ++r)
                rs[r] += __shfl_xor(rs[r], off, 32);
#pragma unroll
        for (int r = 0; r < 8; ++r) lrow[r] = alpha[r] * lrow[r] + rs[r];

        // rescale O, then O += P V
#pragma unroll
        for (int nt = 0; nt < 8; ++nt)
#pragma unroll
            for (int r = 0; r < 8; ++r)
                oacc[nt][r] *= alpha[r];
#pragma unroll
        for (int kc = 0; kc < 4; ++kc) {
            v16bf af = load_a_frag(Pw, 136, nl, kc * 32 + half * 16);
#pragma unroll
            for (int nt = 0; nt < 8; ++nt) {
                v16bf bf = load_b_frag(Vl, 136, nt * 16 + nl, kc * 32 + half * 16);
                oacc[nt] = wmma_bf16(af, bf, oacc[nt]);
            }
        }
    }

    // epilogue: O /= l, store bf16 [B,S,H,HD]
#pragma unroll
    for (int r = 0; r < 8; ++r) lrow[r] = 1.0f / lrow[r];
#pragma unroll
    for (int nt = 0; nt < 8; ++nt)
#pragma unroll
        for (int r = 0; r < 8; ++r) {
            const int qrow = q0 + wave * 16 + r + 8 * half;
            attn[(((size_t)(b * SS + qrow)) * HH + h) * HDD + nt * 16 + nl] =
                f2bf(oacc[nt][r] * lrow[r]);
        }
}

// ---------------------------------------------------------------------------
extern "C" void kernel_launch(void* const* d_in, const int* in_sizes, int n_in,
                              void* d_out, int out_size, void* d_ws, size_t ws_size,
                              hipStream_t stream) {
    const float* x    = (const float*)d_in[0];
    const float* cosb = (const float*)d_in[1];
    const float* sinb = (const float*)d_in[2];
    // d_in[3] = mask (causal, recomputed in-kernel), d_in[8] = start_pos (0)
    const float* wq = (const float*)d_in[4];
    const float* wk = (const float*)d_in[5];
    const float* wv = (const float*)d_in[6];
    const float* wo = (const float*)d_in[7];
    float* out = (float*)d_out;

    char* ws = (char*)d_ws;
    constexpr size_t QF_B  = (size_t)BB * SS * HH * HDD * 4;      // 64 MB
    constexpr size_t KF_B  = (size_t)BB * SS * KVHH * HDD * 4;    // 16 MB
    constexpr size_t QBF_B = (size_t)BB * SS * HH * HDD * 2;      // 32 MB
    constexpr size_t KBF_B = (size_t)BB * SS * KVHH * HDD * 2;    // 8 MB

    float*  qf   = (float*)(ws);
    float*  kf   = (float*)(ws + QF_B);
    float*  vf   = (float*)(ws + QF_B + KF_B);
    __bf16* qbf  = (__bf16*)(ws + QF_B + 2 * KF_B);
    __bf16* kbf  = (__bf16*)(ws + QF_B + 2 * KF_B + QBF_B);
    __bf16* vbf  = (__bf16*)(ws + QF_B + 2 * KF_B + QBF_B + KBF_B);
    __bf16* attn = (__bf16*)(ws);   // reuse q-f32 region after RoPE consumed it

    const dim3 blk(256);
    const int M = BB * SS;  // 4096

    // QKV projections (f32 -> bf16 staging, f32 accumulate out)
    gemm_k<true><<<dim3((HH * HDD) / 128, M / 128), blk, 0, stream>>>(x, wq, qf, M, HH * HDD, DD);
    gemm_k<true><<<dim3((KVHH * HDD) / 128, M / 128), blk, 0, stream>>>(x, wk, kf, M, KVHH * HDD, DD);
    gemm_k<true><<<dim3((KVHH * HDD) / 128, M / 128), blk, 0, stream>>>(x, wv, vf, M, KVHH * HDD, DD);

    // RoPE + repack to bf16
    rope_q_k<<<(BB * SS * HH * 64) / 256, blk, 0, stream>>>(qf, cosb, sinb, qbf);
    rope_kv_k<<<(BB * SS * KVHH * 64) / 256, blk, 0, stream>>>(kf, vf, cosb, sinb, kbf, vbf);

    // flash attention
    const int smem = 4 * 128 * 136 * 2;  // 139264 B
    (void)hipFuncSetAttribute((const void*)attn_k,
                              hipFuncAttributeMaxDynamicSharedMemorySize, smem);
    attn_k<<<dim3(SS / 128, BB * HH), blk, smem, stream>>>(qbf, kbf, vbf, attn);

    // output projection (bf16 A, f32 out)
    gemm_k<false><<<dim3(DD / 128, M / 128), blk, 0, stream>>>(attn, wo, out, M, DD, DD);
}
